// R_77429670412896
// MI455X (gfx1250) — matrix operand order repaired and checked by reading, not compiled, not run on hardware
//
#include <hip/hip_runtime.h>
#include <hip/hip_bf16.h>
#include <cstdint>

// ---------------------------------------------------------------------------
// CDNA5 (gfx1250) wave32 WMMA types and helpers
// ---------------------------------------------------------------------------
typedef __attribute__((ext_vector_type(16))) __bf16 v16bf;
typedef __attribute__((ext_vector_type(8)))  float  v8f;

#define EPSV 1e-7f

union AOp {
    unsigned u[8];
    uint4    q[2];
    v16bf    v;
};

__device__ __forceinline__ unsigned short f2bf(float f) {
    unsigned u = __float_as_uint(f);
    u += 0x7fffu + ((u >> 16) & 1u);       // round-to-nearest-even
    return (unsigned short)(u >> 16);
}
__device__ __forceinline__ float bf2f(unsigned short h) {
    return __uint_as_float(((unsigned)h) << 16);
}
__device__ __forceinline__ unsigned pack2(float lo, float hi) {
    return (unsigned)f2bf(lo) | ((unsigned)f2bf(hi) << 16);
}
__device__ __forceinline__ v8f vzero() {
    v8f z;
#pragma unroll
    for (int k = 0; k < 8; ++k) z[k] = 0.f;
    return z;
}

// A-operand (16x32 bf16) slot->K mapping (half=0), contiguous order:
// lanes 0-15: K=0..15 (v0..7); lanes 16-31: K=16..31
__device__ __forceinline__ int a_kl(int lane, int v) {
    return ((lane >> 4) & 1) * 16 + (v << 1);
}

// Build 16x32 bf16 A-operand in registers from a row-major bf16 matrix.
__device__ __forceinline__ AOp load_a_rm(const unsigned short* __restrict__ X,
                                         long row0, int ld, int k0, int lane) {
    AOp a;
    const int m = lane & 15;
    const unsigned short* rp = X + (row0 + m) * (long)ld + k0;
#pragma unroll
    for (int v = 0; v < 8; ++v)
        a.u[v] = *(const unsigned*)(rp + a_kl(lane, v));  // 2 consecutive bf16
    return a;
}

// Load 32x16 bf16 B-operand from a pre-swizzled operand image
// (tile = 256 u32; lane's 32B are contiguous -> two b128 loads).
__device__ __forceinline__ AOp load_b_img(const unsigned* __restrict__ t, int lane) {
    AOp b;
    const uint4* p = (const uint4*)(t + lane * 8);
    b.q[0] = p[0];
    b.q[1] = p[1];
    return b;
}

__device__ __forceinline__ v8f wmma_bf16(const AOp& a, const AOp& b, v8f c) {
    return __builtin_amdgcn_wmma_f32_16x16x32_bf16(
        /*neg_a=*/false, a.v, /*neg_b=*/false, b.v,
        /*c_mod=*/(short)0, c, /*reuse_a=*/false, /*reuse_b=*/false);
}

// Pipelined 8-tile WMMA sweep: B tile bt+1 is fetched while WMMA bt executes.
__device__ __forceinline__ void wmma_sweep8(const AOp& xa,
                                            const unsigned* __restrict__ tbase,
                                            int lane, v8f* acc) {
    AOp wb = load_b_img(tbase, lane);
#pragma unroll
    for (int bt = 0; bt < 8; ++bt) {
        AOp nb;
        if (bt < 7) nb = load_b_img(tbase + (bt + 1) * 256, lane);
        acc[bt] = wmma_bf16(xa, wb, acc[bt]);
        if (bt < 7) wb = nb;
    }
}

// ---------------------------------------------------------------------------
// Weight repack: W(a,b,c) fp32 -> bf16 B-operand images [c][ach][bt][lane][16]
// ---------------------------------------------------------------------------
__global__ void pack_b_kernel(const float* __restrict__ W,
                              unsigned short* __restrict__ dst,
                              int K, int NC, int total) {
    int idx = blockIdx.x * blockDim.x + threadIdx.x;
    if (idx >= total) return;
    int a   = idx / (128 * NC);
    int rem = idx - a * 128 * NC;
    int b   = rem / NC;
    int c   = rem - b * NC;
    int ach = a >> 5, Kl = a & 31;
    int lanep = ((Kl >> 4) & 1) * 16 + (b & 15);
    int vp = (Kl & 15) >> 1, hp = Kl & 1;
    int bt = b >> 4;
    int ACH = K >> 5;
    unsigned o = (((unsigned)(c * ACH + ach) * 8u + (unsigned)bt) * 256u +
                  (unsigned)lanep * 8u + (unsigned)vp) * 2u + (unsigned)hp;
    dst[o] = f2bf(W[idx]);
}

__global__ void cvt_bf16_kernel(const float* __restrict__ src,
                                unsigned short* __restrict__ dst, int total) {
    int i = blockIdx.x * blockDim.x + threadIdx.x;
    if (i < total) dst[i] = f2bf(src[i]);
}

// ---------------------------------------------------------------------------
// Fused relational conv layer (one workgroup = one batch, 8 waves):
//   Xout[s,i,0:128]   = relu( An_flat(128x512) @ Y(512x128) + Xin@R )
//   Xout[s,i,128:192] = x[s,i,:] (bf16 passthrough)
// Y[j*4+c, b] = (Xin @ W_c)[j,b];  An_flat[i, j*4+c] = A[s,i,j,c]*norm[i,c]
// ---------------------------------------------------------------------------
template <int CIN>
__global__ __launch_bounds__(256) void conv_kernel(
    const float* __restrict__ A, const unsigned short* __restrict__ Xin,
    const unsigned* __restrict__ Wp, const unsigned* __restrict__ Rp,
    const unsigned short* __restrict__ xbf, unsigned short* __restrict__ Xout) {
    extern __shared__ char smem[];
    unsigned* YIMG = (unsigned*)smem;                    // 16 kc * 8 bt * 256 u32 = 128KB
    float* normS   = (float*)(smem + 16 * 8 * 256 * 4);  // 512 f32 = 2KB

    const int  t = threadIdx.x, lane = t & 31, wave = t >> 5;
    const int  s = blockIdx.x;
    const long XROW = (long)s * 128;
    constexpr int ACH = CIN / 32;

    normS[t]       = EPSV;
    normS[t + 256] = EPSV;
    __syncthreads();

    // ---- phase 0: row sums over j per (i,c) via LDS float atomics ----
    {
        int i = t >> 1, jh = t & 1;
        const float* ap = A + (((long)s * 128 + i) * 128 + jh * 64) * 5;
        float s0 = 0.f, s1 = 0.f, s2 = 0.f, s3 = 0.f;
        for (int j = 0; j < 64; j += 4) {
            __builtin_prefetch(ap + 80, 0, 1);  // 16 rows (4 iterations) ahead of this lane
#pragma unroll
            for (int q = 0; q < 4; ++q) {
                s0 += ap[0]; s1 += ap[1]; s2 += ap[2]; s3 += ap[3];
                ap += 5;
            }
        }
        atomicAdd(&normS[i * 4 + 0], s0);
        atomicAdd(&normS[i * 4 + 1], s1);
        atomicAdd(&normS[i * 4 + 2], s2);
        atomicAdd(&normS[i * 4 + 3], s3);
    }

    // ---- phase 1: Y_c = Xin @ W_c, scattered into LDS B-operand images ----
    {
        const int  jt = wave;
        const long j0 = XROW + jt * 16;
        const int  h = (lane >> 4) & 1, n = lane & 15;
        AOp xas[ACH];  // hoist Xin A-tiles once, reuse across all 4 relations
#pragma unroll
        for (int ach = 0; ach < ACH; ++ach)
            xas[ach] = load_a_rm(Xin, j0, CIN, ach * 32, lane);
        for (int c = 0; c < 4; ++c) {
            v8f y[8];
#pragma unroll
            for (int bt = 0; bt < 8; ++bt) y[bt] = vzero();
#pragma unroll
            for (int ach = 0; ach < ACH; ++ach)
                wmma_sweep8(xas[ach], Wp + ((c * ACH + ach) * 8) * 256, lane, y);
            // D layout: value r at (M = r + 8h, N = n); scatter to K = j*4+c image
#pragma unroll
            for (int bt = 0; bt < 8; ++bt) {
#pragma unroll
                for (int r = 0; r < 8; ++r) {
                    int j  = jt * 16 + r + 8 * h;
                    int K  = j * 4 + c;
                    int kc = K >> 5, Kl = K & 31;
                    int lp = ((Kl >> 4) & 1) * 16 + n;
                    int vp = (Kl & 15) >> 1, hp = Kl & 1;
                    unsigned idx = (unsigned)((kc * 8 + bt) * 256 + lp * 8 + vp);
                    ((unsigned short*)YIMG)[idx * 2 + hp] = f2bf(y[bt][r]);
                }
            }
        }
    }
    __syncthreads();
    normS[t]       = 1.f / normS[t];
    normS[t + 256] = 1.f / normS[t + 256];
    __syncthreads();

    // ---- phase 2: Out = Xin@R + An_flat @ Y ----
    const int i0 = wave * 16;
    const int m = lane & 15, h = (lane >> 4) & 1, n = lane & 15;
    v8f acc[8];
#pragma unroll
    for (int bt = 0; bt < 8; ++bt) acc[bt] = vzero();

#pragma unroll
    for (int ach = 0; ach < ACH; ++ach) {  // residual term Xin @ R
        AOp xa = load_a_rm(Xin, XROW + i0, CIN, ach * 32, lane);
        wmma_sweep8(xa, Rp + (ach * 8) * 256, lane, acc);
    }

    // K-loop over An_flat: c = Kl&3 is kc-invariant -> hoist norms & pointers;
    // stream advances by 8 rows (40 floats) per kc. Two-stage software pipeline.
    const float* Abase = A + ((long)s * 128 + i0 + m) * 128 * 5;
    const float* pA[8];
    float2       nr[8];
#pragma unroll
    for (int v = 0; v < 8; ++v) {
        int Kl = a_kl(lane, v);
        pA[v] = Abase + (Kl >> 2) * 5 + (Kl & 3);
        nr[v] = *(const float2*)&normS[(i0 + m) * 4 + (Kl & 3)];
    }
    float2 cur[8];
#pragma unroll
    for (int v = 0; v < 8; ++v) { cur[v].x = pA[v][0]; cur[v].y = pA[v][1]; }

    for (int kc = 0; kc < 16; ++kc) {
        __builtin_prefetch(pA[0] + (kc + 2) * 40, 0, 1);  // A stream prefetch
        float2 nxt[8];
        if (kc < 15) {
#pragma unroll
            for (int v = 0; v < 8; ++v) {
                const float* p = pA[v] + (kc + 1) * 40;
                nxt[v].x = p[0];
                nxt[v].y = p[1];
            }
        }
        AOp an;
#pragma unroll
        for (int v = 0; v < 8; ++v)
            an.u[v] = pack2(cur[v].x * nr[v].x, cur[v].y * nr[v].y);
        wmma_sweep8(an, YIMG + (kc * 8) * 256, lane, acc);
        if (kc < 15) {
#pragma unroll
            for (int v = 0; v < 8; ++v) cur[v] = nxt[v];
        }
    }

    // ---- epilogue: relu, bf16 store, x passthrough ----
#pragma unroll
    for (int bt = 0; bt < 8; ++bt) {
#pragma unroll
        for (int r = 0; r < 8; ++r) {
            float val = fmaxf(acc[bt][r], 0.f);
            long  row = XROW + i0 + r + 8 * h;
            Xout[row * 192 + bt * 16 + n] = f2bf(val);
        }
    }
    {
        long row  = XROW + i0 + (lane >> 1);
        int  half = lane & 1;
        const unsigned* src = (const unsigned*)(xbf + row * 64 + half * 32);
        unsigned*       dst = (unsigned*)(Xout + row * 192 + 128 + half * 32);
#pragma unroll
        for (int k = 0; k < 16; ++k) dst[k] = src[k];
    }
}

// ---------------------------------------------------------------------------
// Readout: gate softmax over nodes, tanh features via WMMA, gated pool,
// linear head, sigmoid. One workgroup per batch.
// ---------------------------------------------------------------------------
__global__ __launch_bounds__(256) void readout_kernel(
    const unsigned short* __restrict__ X3, const unsigned* __restrict__ NNWp,
    const float* __restrict__ gate_w, const float* __restrict__ gate_b,
    const float* __restrict__ nn_b, const float* __restrict__ lin_w,
    const float* __restrict__ lin_b, float* __restrict__ out) {
    __shared__ float sg[128], sgate[128], shg[128];
    __shared__ float smax, ssum, sscal;
    const int  t = threadIdx.x, lane = t & 31, wave = t >> 5;
    const int  s = blockIdx.x;
    const long XROW = (long)s * 128;

    if (t < 128) {  // gate logits
        const unsigned short* xp = X3 + (XROW + t) * 192;
        float d = gate_b[0];
        for (int k = 0; k < 192; k += 2) {
            unsigned u = *(const unsigned*)(xp + k);
            d += bf2f((unsigned short)(u & 0xffffu)) * gate_w[k] +
                 bf2f((unsigned short)(u >> 16)) * gate_w[k + 1];
        }
        sg[t] = d;
    }
    __syncthreads();
    if (t == 0) {
        float mx = sg[0];
        for (int i = 1; i < 128; ++i) mx = fmaxf(mx, sg[i]);
        smax = mx; ssum = 0.f; sscal = lin_b[0];
    }
    __syncthreads();
    if (t < 128) {
        float e = __expf(sg[t] - smax);
        sg[t] = e;
        atomicAdd(&ssum, e);
        shg[t] = 0.f;
    }
    __syncthreads();
    if (t < 128) sgate[t] = sg[t] / ssum;
    __syncthreads();

    {  // feat = tanh(X3 @ nn_w + nn_b); gated pool into shg
        const int i0 = wave * 16, h = (lane >> 4) & 1, n = lane & 15;
        float ga[8];
#pragma unroll
        for (int r = 0; r < 8; ++r) ga[r] = sgate[i0 + r + 8 * h];
        v8f acc[8];
#pragma unroll
        for (int bt = 0; bt < 8; ++bt) acc[bt] = vzero();
#pragma unroll
        for (int ach = 0; ach < 6; ++ach) {
            AOp xa = load_a_rm(X3, XROW + i0, 192, ach * 32, lane);
            wmma_sweep8(xa, NNWp + (ach * 8) * 256, lane, acc);
        }
#pragma unroll
        for (int bt = 0; bt < 8; ++bt) {
            float bb = nn_b[bt * 16 + n];
            float local = 0.f;
#pragma unroll
            for (int r = 0; r < 8; ++r) local += ga[r] * tanhf(acc[bt][r] + bb);
            atomicAdd(&shg[bt * 16 + n], local);
        }
    }
    __syncthreads();
    if (t < 128) atomicAdd(&sscal, fmaxf(shg[t], 0.f) * lin_w[t]);
    __syncthreads();
    if (t == 0) {
        float v = fmaxf(sscal, 0.f);
        out[s] = 1.f / (1.f + __expf(-v));
    }
}

// ---------------------------------------------------------------------------
extern "C" void kernel_launch(void* const* d_in, const int* in_sizes, int n_in,
                              void* d_out, int out_size, void* d_ws, size_t ws_size,
                              hipStream_t stream) {
    (void)in_sizes; (void)n_in; (void)out_size; (void)ws_size;
    const float* A      = (const float*)d_in[0];
    const float* x      = (const float*)d_in[1];
    const float* w1     = (const float*)d_in[2];
    const float* r1     = (const float*)d_in[3];
    const float* w2     = (const float*)d_in[4];
    const float* r2     = (const float*)d_in[5];
    const float* gate_w = (const float*)d_in[6];
    const float* gate_b = (const float*)d_in[7];
    const float* nn_w   = (const float*)d_in[8];
    const float* nn_b   = (const float*)d_in[9];
    const float* lin_w  = (const float*)d_in[10];
    const float* lin_b  = (const float*)d_in[11];
    float*       out    = (float*)d_out;

    char*  ws  = (char*)d_ws;
    size_t off = 0;
    auto alloc = [&](size_t bytes) {
        char* p = ws + off;
        off += (bytes + 255) & ~(size_t)255;
        return p;
    };
    unsigned short* xbf = (unsigned short*)alloc(512ull * 128 * 64 * 2);
    unsigned short* X1  = (unsigned short*)alloc(512ull * 128 * 192 * 2);
    unsigned short* X2  = (unsigned short*)alloc(512ull * 128 * 192 * 2);
    unsigned* W1p = (unsigned*)alloc(4 * 2 * 8 * 1024);
    unsigned* R1p = (unsigned*)alloc(2 * 8 * 1024);
    unsigned* W2p = (unsigned*)alloc(4 * 6 * 8 * 1024);
    unsigned* R2p = (unsigned*)alloc(6 * 8 * 1024);
    unsigned* NWp = (unsigned*)alloc(6 * 8 * 1024);

    {
        int total = 512 * 128 * 64;
        cvt_bf16_kernel<<<(total + 255) / 256, 256, 0, stream>>>(x, xbf, total);
    }
    pack_b_kernel<<<(64 * 128 * 4 + 255) / 256, 256, 0, stream>>>(w1, (unsigned short*)W1p, 64, 4, 64 * 128 * 4);
    pack_b_kernel<<<(64 * 128 + 255) / 256, 256, 0, stream>>>(r1, (unsigned short*)R1p, 64, 1, 64 * 128);
    pack_b_kernel<<<(192 * 128 * 4 + 255) / 256, 256, 0, stream>>>(w2, (unsigned short*)W2p, 192, 4, 192 * 128 * 4);
    pack_b_kernel<<<(192 * 128 + 255) / 256, 256, 0, stream>>>(r2, (unsigned short*)R2p, 192, 1, 192 * 128);
    pack_b_kernel<<<(192 * 128 + 255) / 256, 256, 0, stream>>>(nn_w, (unsigned short*)NWp, 192, 1, 192 * 128);

    const int SMEM = 16 * 8 * 256 * 4 + 512 * 4;  // 128KB Y images + 2KB norms
    hipFuncSetAttribute(reinterpret_cast<const void*>(conv_kernel<64>),
                        hipFuncAttributeMaxDynamicSharedMemorySize, SMEM);
    hipFuncSetAttribute(reinterpret_cast<const void*>(conv_kernel<192>),
                        hipFuncAttributeMaxDynamicSharedMemorySize, SMEM);

    conv_kernel<64><<<512, 256, SMEM, stream>>>(A, xbf, W1p, R1p, xbf, X1);
    conv_kernel<192><<<512, 256, SMEM, stream>>>(A, X1, W2p, R2p, xbf, X2);
    readout_kernel<<<512, 256, 0, stream>>>(X2, NWp, gate_w, gate_b, nn_b, lin_w,
                                            lin_b, out);
}
